// DRLPCR_GNN_25546465477020
// MI455X (gfx1250) — matrix operand order repaired and checked by reading, not compiled, not run on hardware
//
#include <hip/hip_runtime.h>

typedef __attribute__((ext_vector_type(16))) _Float16 v16h;
typedef __attribute__((ext_vector_type(8)))  _Float16 v8h;
typedef __attribute__((ext_vector_type(8)))  float    v8f;

constexpr int PN = 200000;   // num paths
constexpr int CN = 50000;    // num channels
constexpr int KN = 8;        // neighbors per path
constexpr int FN = 32;       // feature dim
constexpr int NITER = 5;
constexpr int LDS_STRIDE_H = 40;  // f16 units; 16B-aligned rows, conflict-free b128 reads

// ws layout (bytes): [agg: CN*FN*4][hc16: CN*FN*2][wpack: 4*6*32*16*2]
constexpr size_t WS_AGG_OFF   = 0;
constexpr size_t WS_HC16_OFF  = (size_t)CN * FN * 4;                // 6,400,000
constexpr size_t WS_WPACK_OFF = WS_HC16_OFF + (size_t)CN * FN * 2;  // 9,600,000 (32B aligned)
constexpr int    WPACK_MAT    = 6 * 32 * 16;  // f16 elems per packed weight matrix

__device__ __forceinline__ v8f wmma_f16(v16h a, v16h b, v8f c) {
  // D = A*B + C, 16x16x32, f16 in / f32 accum  (v_wmma_f32_16x16x32_f16)
  return __builtin_amdgcn_wmma_f32_16x16x32_f16(false, a, false, b, (short)0, c,
                                                false, false);
}

// Fast activations. CDNA5 has a hardware V_TANH_F32 (ISA sec 8.4 transcendental
// list); use it when the builtin exists, else exp+rcp (no IEEE divides).
#if __has_builtin(__builtin_amdgcn_tanhf)
__device__ __forceinline__ float tanh_fast(float x) {
  return __builtin_amdgcn_tanhf(x);
}
__device__ __forceinline__ float sigm(float x) {
  return __builtin_fmaf(0.5f, __builtin_amdgcn_tanhf(0.5f * x), 0.5f);
}
#else
__device__ __forceinline__ float tanh_fast(float x) {
  // 1 - 2/(e^{2x}+1); saturation-safe: exp->inf => rcp->0 => 1, exp->0 => -1.
  return 1.0f - 2.0f * __builtin_amdgcn_rcpf(__expf(2.0f * x) + 1.0f);
}
__device__ __forceinline__ float sigm(float x) {
  return __builtin_amdgcn_rcpf(1.0f + __expf(-x));
}
#endif

// A operand slice from an f16 row of 32 elements (row-major).
// Lane holds row L%16; half = L/16: elems 0-7 <- K = 8h..8h+7, elems 8-15 <- K = 16+8h..
__device__ __forceinline__ v16h load_A_f16row(const _Float16* row, int half) {
  v8h a = *reinterpret_cast<const v8h*>(row + 8 * half);
  v8h b = *reinterpret_cast<const v8h*>(row + 16 + 8 * half);
  v16h r;
#pragma unroll
  for (int e = 0; e < 8; ++e) { r[e] = a[e]; r[8 + e] = b[e]; }
  return r;
}

// A operand slice from an fp32 row of 32 elements (used for agg, which must stay fp32).
__device__ __forceinline__ v16h load_A_f32row(const float* row, int half) {
  const float4* lo = reinterpret_cast<const float4*>(row + 8 * half);
  const float4* hi = reinterpret_cast<const float4*>(row + 16 + 8 * half);
  float4 a0 = lo[0], a1 = lo[1], b0 = hi[0], b1 = hi[1];
  v16h r;
  r[0]  = (_Float16)a0.x; r[1]  = (_Float16)a0.y; r[2]  = (_Float16)a0.z; r[3]  = (_Float16)a0.w;
  r[4]  = (_Float16)a1.x; r[5]  = (_Float16)a1.y; r[6]  = (_Float16)a1.z; r[7]  = (_Float16)a1.w;
  r[8]  = (_Float16)b0.x; r[9]  = (_Float16)b0.y; r[10] = (_Float16)b0.z; r[11] = (_Float16)b0.w;
  r[12] = (_Float16)b1.x; r[13] = (_Float16)b1.y; r[14] = (_Float16)b1.z; r[15] = (_Float16)b1.w;
  return r;
}

// One GRU cell for a 16-row tile, 12 WMMAs, one 16-col gate group at a time so
// only 4 accumulator tiles (32 VGPRs) stay live.
// hC[2] = fp32 state in C layout (VGPR i -> row i+8*half, col = t*16 + lane%16).
__device__ __forceinline__ void gru_tile(v16h xA, v16h hA,
                                         const v16h Bih[6], const v16h Bhh[6],
                                         const float brz[4], const float bin[2],
                                         const float bhn[2], v8f hC[2]) {
#pragma unroll
  for (int t = 0; t < 2; ++t) {
    v8f ar, az, ai, ah;
#pragma unroll
    for (int i = 0; i < 8; ++i) {
      ar[i] = brz[t];
      az[i] = brz[2 + t];
      ai[i] = bin[t];
      ah[i] = bhn[t];
    }
    ar = wmma_f16(xA, Bih[t], wmma_f16(hA, Bhh[t], ar));           // r pre-act
    az = wmma_f16(xA, Bih[2 + t], wmma_f16(hA, Bhh[2 + t], az));   // z pre-act
    ai = wmma_f16(xA, Bih[4 + t], ai);                             // i_n
    ah = wmma_f16(hA, Bhh[4 + t], ah);                             // h_n
#pragma unroll
    for (int i = 0; i < 8; ++i) {
      float r = sigm(ar[i]);
      float z = sigm(az[i]);
      float nn = tanh_fast(ai[i] + r * ah[i]);
      hC[t][i] = nn + z * (hC[t][i] - nn);  // (1-z)*n + z*h
    }
  }
}

__device__ __forceinline__ void load_weight_tiles(const _Float16* __restrict__ pack,
                                                  int lane, v16h B[6]) {
  const v16h* p = reinterpret_cast<const v16h*>(pack);
#pragma unroll
  for (int t = 0; t < 6; ++t) B[t] = p[t * 32 + lane];
}

// ---------------- path update: 8 GRU1 steps per path + scatter-add ----------------
__global__ __launch_bounds__(128) void path_update_kernel(
    const int* __restrict__ p2c, const _Float16* __restrict__ hc16,
    float* __restrict__ h_p, const _Float16* __restrict__ wih_pack,
    const _Float16* __restrict__ whh_pack, const float* __restrict__ bih,
    const float* __restrict__ bhh, float* __restrict__ agg) {
  __shared__ __align__(16) _Float16 xpose[4][16 * LDS_STRIDE_H];
  const int lane = threadIdx.x & 31;
  const int wv   = threadIdx.x >> 5;
  const int n    = lane & 15;
  const int half = lane >> 4;
  const int m0   = (blockIdx.x * 4 + wv) * 16;

  v16h Bih6[6], Bhh6[6];
  load_weight_tiles(wih_pack, lane, Bih6);
  load_weight_tiles(whh_pack, lane, Bhh6);

  float brz[4], bin[2], bhn[2];
#pragma unroll
  for (int t = 0; t < 4; ++t) brz[t] = bih[t * 16 + n] + bhh[t * 16 + n];
#pragma unroll
  for (int t = 0; t < 2; ++t) {
    bin[t] = bih[64 + t * 16 + n];
    bhn[t] = bhh[64 + t * 16 + n];
  }

  const int* ip = p2c + (m0 + n) * KN;
  _Float16* lds = &xpose[wv][0];

  // fp32 master state (C layout) + f16 A operand of h.
  v8f hC[2];
#pragma unroll
  for (int t = 0; t < 2; ++t)
#pragma unroll
    for (int i = 0; i < 8; ++i)
      hC[t][i] = h_p[(m0 + i + 8 * half) * FN + t * 16 + n];
  v16h hA = load_A_f32row(h_p + (m0 + n) * FN, half);

  v16h xA = load_A_f16row(hc16 + (size_t)ip[0] * FN, half);
#pragma unroll 1
  for (int k = 0; k < KN; ++k) {
    // Issue next gather early (redundant reload of last index on final step).
    int kn = (k + 1 < KN) ? (k + 1) : (KN - 1);
    v16h xN = load_A_f16row(hc16 + (size_t)ip[kn] * FN, half);

    gru_tile(xA, hA, Bih6, Bhh6, brz, bin, bhn, hC);

    // C layout -> A layout for the next step via per-wave LDS transpose (f16).
    // Per-wave private region + in-order LDS ops => no barrier needed.
#pragma unroll
    for (int t = 0; t < 2; ++t)
#pragma unroll
      for (int i = 0; i < 8; ++i)
        lds[(i + 8 * half) * LDS_STRIDE_H + t * 16 + n] = (_Float16)hC[t][i];
    hA = load_A_f16row(lds + n * LDS_STRIDE_H, half);
    xA = xN;
  }

  // Write final path state and scatter-add into per-channel aggregation.
#pragma unroll
  for (int t = 0; t < 2; ++t)
#pragma unroll
    for (int i = 0; i < 8; ++i)
      h_p[(m0 + i + 8 * half) * FN + t * 16 + n] = hC[t][i];

#pragma unroll
  for (int k = 0; k < KN; ++k)
#pragma unroll
    for (int i = 0; i < 8; ++i) {
      int c = p2c[(m0 + i + 8 * half) * KN + k];
#pragma unroll
      for (int t = 0; t < 2; ++t)
        atomicAdd(&agg[c * FN + t * 16 + n], hC[t][i]);
    }
}

// ---------------- channel update: h_c = GRU2(agg, h_c) ----------------
__global__ __launch_bounds__(160) void channel_update_kernel(
    const float* __restrict__ agg, float* __restrict__ h_c,
    const _Float16* __restrict__ hc16, const _Float16* __restrict__ wih_pack,
    const _Float16* __restrict__ whh_pack, const float* __restrict__ bih,
    const float* __restrict__ bhh) {
  const int lane = threadIdx.x & 31;
  const int wv   = threadIdx.x >> 5;
  const int n    = lane & 15;
  const int half = lane >> 4;
  const int m0   = (blockIdx.x * 5 + wv) * 16;

  v16h Bih6[6], Bhh6[6];
  load_weight_tiles(wih_pack, lane, Bih6);
  load_weight_tiles(whh_pack, lane, Bhh6);

  float brz[4], bin[2], bhn[2];
#pragma unroll
  for (int t = 0; t < 4; ++t) brz[t] = bih[t * 16 + n] + bhh[t * 16 + n];
#pragma unroll
  for (int t = 0; t < 2; ++t) {
    bin[t] = bih[64 + t * 16 + n];
    bhn[t] = bhh[64 + t * 16 + n];
  }

  v8f hC[2];
#pragma unroll
  for (int t = 0; t < 2; ++t)
#pragma unroll
    for (int i = 0; i < 8; ++i)
      hC[t][i] = h_c[(m0 + i + 8 * half) * FN + t * 16 + n];
  v16h hA = load_A_f16row(hc16 + (size_t)(m0 + n) * FN, half);  // f16 mirror of h_c
  v16h xA = load_A_f32row(agg + (size_t)(m0 + n) * FN, half);   // agg stays fp32

  gru_tile(xA, hA, Bih6, Bhh6, brz, bin, bhn, hC);

#pragma unroll
  for (int t = 0; t < 2; ++t)
#pragma unroll
    for (int i = 0; i < 8; ++i)
      h_c[(m0 + i + 8 * half) * FN + t * 16 + n] = hC[t][i];
}

// ---------------- one-time / per-iteration prep ----------------
__global__ void pack_weights_kernel(const float* __restrict__ W0,
                                    const float* __restrict__ W1,
                                    const float* __restrict__ W2,
                                    const float* __restrict__ W3,
                                    _Float16* __restrict__ wpack) {
  int tid = blockIdx.x * blockDim.x + threadIdx.x;
  if (tid >= 4 * 6 * 32) return;
  int m = tid / 192, r = tid % 192, t = r / 32, lane = r % 32;
  int n = lane & 15, kb = (lane >> 4) * 8;
  const float* W = (m == 0) ? W0 : (m == 1) ? W1 : (m == 2) ? W2 : W3;
  const float* src = W + (t * 16 + n) * FN + kb;
  _Float16* dst = wpack + (size_t)tid * 16;
#pragma unroll
  for (int e = 0; e < 16; ++e) dst[e] = (_Float16)src[(e < 8) ? e : e + 8];
}

__global__ void init_state_kernel(const float* __restrict__ paths,
                                  const float* __restrict__ channels,
                                  float* __restrict__ hp, float* __restrict__ hc) {
  int i = blockIdx.x * blockDim.x + threadIdx.x;
  if (i < PN * FN) hp[i] = paths[i];
  if (i < CN * FN) hc[i] = channels[i];
}

// Fused per-iteration prep: zero agg + refresh f16 mirror of h_c.
__global__ void prep_iter_kernel(const float* __restrict__ h_c,
                                 _Float16* __restrict__ hc16,
                                 float* __restrict__ agg) {
  int i = blockIdx.x * blockDim.x + threadIdx.x;
  if (i < CN * FN) {
    hc16[i] = (_Float16)h_c[i];
    agg[i] = 0.0f;
  }
}

extern "C" void kernel_launch(void* const* d_in, const int* in_sizes, int n_in,
                              void* d_out, int out_size, void* d_ws, size_t ws_size,
                              hipStream_t stream) {
  const float* paths    = (const float*)d_in[0];
  const float* channels = (const float*)d_in[1];
  const int*   p2c      = (const int*)d_in[2];
  const float* Wih1     = (const float*)d_in[3];
  const float* Whh1     = (const float*)d_in[4];
  const float* bih1     = (const float*)d_in[5];
  const float* bhh1     = (const float*)d_in[6];
  const float* Wih2     = (const float*)d_in[7];
  const float* Whh2     = (const float*)d_in[8];
  const float* bih2     = (const float*)d_in[9];
  const float* bhh2     = (const float*)d_in[10];

  float* h_p = (float*)d_out;                  // [P, F]
  float* h_c = h_p + (size_t)PN * FN;          // [C, F]

  char* ws = (char*)d_ws;
  float*     agg   = (float*)(ws + WS_AGG_OFF);       // [C, F] fp32
  _Float16*  hc16  = (_Float16*)(ws + WS_HC16_OFF);   // [C, F] f16 mirror
  _Float16*  wpack = (_Float16*)(ws + WS_WPACK_OFF);  // packed f16 weights

  const _Float16* wih1p = wpack + 0 * WPACK_MAT;
  const _Float16* whh1p = wpack + 1 * WPACK_MAT;
  const _Float16* wih2p = wpack + 2 * WPACK_MAT;
  const _Float16* whh2p = wpack + 3 * WPACK_MAT;

  pack_weights_kernel<<<3, 256, 0, stream>>>(Wih1, Whh1, Wih2, Whh2, wpack);

  const int initN = PN * FN;
  init_state_kernel<<<(initN + 255) / 256, 256, 0, stream>>>(paths, channels, h_p, h_c);

  const int cElems = CN * FN;
  for (int it = 0; it < NITER; ++it) {
    prep_iter_kernel<<<(cElems + 255) / 256, 256, 0, stream>>>(h_c, hc16, agg);
    path_update_kernel<<<PN / 64, 128, 0, stream>>>(p2c, hc16, h_p, wih1p, whh1p,
                                                    bih1, bhh1, agg);
    channel_update_kernel<<<CN / 80, 160, 0, stream>>>(agg, h_c, hc16, wih2p,
                                                       whh2p, bih2, bhh2);
  }
}
